// BiLSTM_63127429316858
// MI455X (gfx1250) — compile-verified, matrix operand order checked by reading, not run on hardware
//
#include <hip/hip_runtime.h>
#include <math.h>

// Problem constants
#define Vv 30000
#define Hh 256
#define Tt 9
#define Bb 64
#define Ss 512
#define G4H 1024   // 4*H
#define KC  512    // 2*H (x_t concat h)

typedef __bf16 bf16;
typedef __attribute__((ext_vector_type(16))) __bf16 v16bf;
typedef __attribute__((ext_vector_type(8)))  __bf16 v8bf;
typedef __attribute__((ext_vector_type(8)))  float  v8f;

union Op32 { v16bf v; v8bf h[2]; };

// ---------------- prep kernels ----------------

// xsb[t][b][h] = bf16(emb[x[b][t]][h])   (time-major for the LSTM A-operand)
__global__ void k_embed(const int* __restrict__ x, const float* __restrict__ emb,
                        bf16* __restrict__ xsb) {
  int idx = blockIdx.x * blockDim.x + threadIdx.x;     // over S*B*H = 2^23
  if (idx >= Ss * Bb * Hh) return;
  int h = idx & (Hh - 1);
  int b = (idx >> 8) & (Bb - 1);
  int t = idx >> 14;
  int tok = x[b * Ss + t];
  xsb[idx] = (bf16)emb[(size_t)tok * Hh + h];
}

// wb[dir][g][k] bf16: k<256 -> w_ih[g][k], else w_hh[g][k-256]
__global__ void k_pack_w(const float* __restrict__ wihf, const float* __restrict__ whhf,
                         const float* __restrict__ wihb, const float* __restrict__ whhb,
                         bf16* __restrict__ wb) {
  int idx = blockIdx.x * blockDim.x + threadIdx.x;     // 2*1024*512
  if (idx >= 2 * G4H * KC) return;
  int k = idx & (KC - 1);
  int g = (idx >> 9) & (G4H - 1);
  int d = idx >> 19;
  const float* wih = d ? wihb : wihf;
  const float* whh = d ? whhb : whhf;
  float v = (k < Hh) ? wih[g * Hh + k] : whh[g * Hh + (k - Hh)];
  wb[idx] = (bf16)v;
}

// biasc[dir][g] = b_ih + b_hh
__global__ void k_bias(const float* __restrict__ bihf, const float* __restrict__ bhhf,
                       const float* __restrict__ bihb, const float* __restrict__ bhhb,
                       float* __restrict__ bc) {
  int idx = blockIdx.x * blockDim.x + threadIdx.x;     // 2048
  if (idx >= 2 * G4H) return;
  int g = idx & (G4H - 1);
  int d = idx >> 10;
  bc[idx] = d ? (bihb[g] + bhhb[g]) : (bihf[g] + bhhf[g]);
}

// ---------------- persistent bidirectional LSTM ----------------
// grid = 2 workgroups (blockIdx.x = direction), 1024 threads = 32 waves.
// Per step: gates(64x1024) = [x_t | h](64x512) @ Wcat^T via v_wmma_f32_16x16x32_bf16.
// Each wave owns 2 N-tiles x 4 M-tiles (8 accumulators).
// K-loop statically split (no branches); kt kept as a register loop variable
// (unroll 2) so weight loads can NOT be hoisted across the 512-step loop
// (full hoist => 256 VGPRs of weights => scratch spills, seen in round 2).
__global__ __launch_bounds__(1024)
void k_lstm(const bf16* __restrict__ xsb, const bf16* __restrict__ wb,
            const float* __restrict__ biasc,
            bf16* __restrict__ hsf, bf16* __restrict__ hsb) {
  extern __shared__ char smem_raw[];
  float* gates = (float*)smem_raw;                      // 64 x 1024 f32 = 256KB
  bf16*  hb    = (bf16*)(smem_raw + 64 * 1024 * 4);     // 64 x 256 bf16 = 32KB

  const int   dir = blockIdx.x;
  const bf16* w   = wb + (size_t)dir * G4H * KC;
  const float* bc = biasc + dir * G4H;
  bf16* hs        = dir ? hsb : hsf;

  const int tid  = threadIdx.x;
  const int lane = tid & 31;
  const int wv   = tid >> 5;
  const int lo   = lane & 15;     // row/col within tile
  const int hi   = lane >> 4;     // K-half / M-half selector
  const int koff = hi * 8;

  // zero h0 in LDS
  for (int i = tid; i < Bb * Hh; i += 1024) hb[i] = (bf16)0.0f;

  // each thread owns 16 cells: batch cm, cell range [cj, cj+16)
  const int cm = tid >> 4;
  const int cj = (tid & 15) << 4;
  float cst[16];
#pragma unroll
  for (int u = 0; u < 16; ++u) cst[u] = 0.0f;

  __syncthreads();

  const int nt0 = wv * 2, nt1 = wv * 2 + 1;
  // per-wave B bases (k-contiguous packed weights)
  const bf16* wB0 = w + (size_t)(nt0 * 16 + lo) * KC + koff;
  const bf16* wB1 = w + (size_t)(nt1 * 16 + lo) * KC + koff;
  // biases for this wave's two gate columns (folded into gate spill)
  const float bs0 = bc[nt0 * 16 + lo];
  const float bs1 = bc[nt1 * 16 + lo];
  // A bases: global x rows (t-dependent bump), LDS h rows (fixed)
  const bf16* xrow = xsb + (size_t)lo * Hh + koff;   // + t*B*H + mt*16*H + kb
  const bf16* hrow = hb + lo * Hh + koff;            // + mt*16*H + (kb-256)

  for (int step = 0; step < Ss; ++step) {
    const int t = dir ? (Ss - 1 - step) : step;
    const bf16* xt = xrow + (size_t)t * Bb * Hh;

    v8f acc[4][2];
#pragma unroll
    for (int mt = 0; mt < 4; ++mt) {
      acc[mt][0] = (v8f){0.f,0.f,0.f,0.f,0.f,0.f,0.f,0.f};
      acc[mt][1] = (v8f){0.f,0.f,0.f,0.f,0.f,0.f,0.f,0.f};
    }

    // ---- K half 1: A from global x_t ----
#pragma unroll 2
    for (int kt = 0; kt < 8; ++kt) {
      const int kb = kt * 32;
      Op32 b0, b1, a;
      b0.h[0] = *(const v8bf*)(wB0 + kb);
      b0.h[1] = *(const v8bf*)(wB0 + kb + 16);
      b1.h[0] = *(const v8bf*)(wB1 + kb);
      b1.h[1] = *(const v8bf*)(wB1 + kb + 16);
#pragma unroll
      for (int mt = 0; mt < 4; ++mt) {
        const bf16* pa = xt + (size_t)(mt * 16) * Hh + kb;
        a.h[0] = *(const v8bf*)pa;
        a.h[1] = *(const v8bf*)(pa + 16);
        acc[mt][0] = __builtin_amdgcn_wmma_f32_16x16x32_bf16(
            false, a.v, false, b0.v, (short)0, acc[mt][0], false, false);
        acc[mt][1] = __builtin_amdgcn_wmma_f32_16x16x32_bf16(
            false, a.v, false, b1.v, (short)0, acc[mt][1], false, false);
      }
    }
    // ---- K half 2: A from LDS h ----
#pragma unroll 2
    for (int kt = 8; kt < 16; ++kt) {
      const int kb = kt * 32;
      Op32 b0, b1, a;
      b0.h[0] = *(const v8bf*)(wB0 + kb);
      b0.h[1] = *(const v8bf*)(wB0 + kb + 16);
      b1.h[0] = *(const v8bf*)(wB1 + kb);
      b1.h[1] = *(const v8bf*)(wB1 + kb + 16);
#pragma unroll
      for (int mt = 0; mt < 4; ++mt) {
        const bf16* pa = hrow + mt * 16 * Hh + (kb - Hh);
        a.h[0] = *(const v8bf*)pa;
        a.h[1] = *(const v8bf*)(pa + 16);
        acc[mt][0] = __builtin_amdgcn_wmma_f32_16x16x32_bf16(
            false, a.v, false, b0.v, (short)0, acc[mt][0], false, false);
        acc[mt][1] = __builtin_amdgcn_wmma_f32_16x16x32_bf16(
            false, a.v, false, b1.v, (short)0, acc[mt][1], false, false);
      }
    }

    // spill biased gates to LDS (C layout: VGPR r -> M = r + 8*hi, N = lo)
#pragma unroll
    for (int mt = 0; mt < 4; ++mt) {
#pragma unroll
      for (int r = 0; r < 8; ++r) {
        const int row = mt * 16 + hi * 8 + r;
        gates[row * G4H + nt0 * 16 + lo] = acc[mt][0][r] + bs0;
        gates[row * G4H + nt1 * 16 + lo] = acc[mt][1][r] + bs1;
      }
    }
    __syncthreads();

    // prefetch next step's x rows for this thread's batch row (global_prefetch_b8)
    {
      const int tn = dir ? (t - 1) : (t + 1);
      if (tn >= 0 && tn < Ss)
        __builtin_prefetch(xsb + ((size_t)tn * Bb + cm) * Hh + cj, 0, 3);
    }

    // pointwise LSTM cell update for owned cells (bias already folded in)
    bf16 hloc[16];
    const float* gr = gates + cm * G4H;
#pragma unroll
    for (int u = 0; u < 16; ++u) {
      const int j = cj + u;
      float ig = gr[j];
      float fg = gr[Hh + j];
      float gg = gr[2 * Hh + j];
      float og = gr[3 * Hh + j];
      float si = 1.0f / (1.0f + __expf(-ig));
      float sf = 1.0f / (1.0f + __expf(-fg));
      float so = 1.0f / (1.0f + __expf(-og));
      float tg = tanhf(gg);
      cst[u] = sf * cst[u] + si * tg;
      hloc[u] = (bf16)(so * tanhf(cst[u]));
    }
    __syncthreads();   // gates fully consumed; hb reads of this step done

#pragma unroll
    for (int u = 0; u < 16; ++u) hb[cm * Hh + cj + u] = hloc[u];
    {
      bf16* dst = hs + ((size_t)t * Bb + cm) * Hh + cj;
#pragma unroll
      for (int u = 0; u < 16; ++u) dst[u] = hloc[u];
    }
    __syncthreads();   // new h visible before next step's A-loads
  }
}

// ---------------- logits: feat(512) @ fc_w^T + fc_b, T=9 ----------------
__global__ __launch_bounds__(256)
void k_logits(const bf16* __restrict__ hsf, const bf16* __restrict__ hsb,
              const float* __restrict__ fcw, const float* __restrict__ fcb,
              float* __restrict__ logits) {
  __shared__ float sfw[Tt * KC];
  __shared__ float sfb[Tt];
  for (int i = threadIdx.x; i < Tt * KC; i += blockDim.x) sfw[i] = fcw[i];
  if (threadIdx.x < Tt) sfb[threadIdx.x] = fcb[threadIdx.x];
  __syncthreads();

  int pos = blockIdx.x * blockDim.x + threadIdx.x;   // t*64 + b
  if (pos >= Ss * Bb) return;
  const bf16* pf = hsf + (size_t)pos * Hh;
  const bf16* pb = hsb + (size_t)pos * Hh;
  float acc[Tt];
#pragma unroll
  for (int j = 0; j < Tt; ++j) acc[j] = sfb[j];
  for (int h = 0; h < Hh; ++h) {
    float vf = (float)pf[h];
    float vb = (float)pb[h];
#pragma unroll
    for (int j = 0; j < Tt; ++j)
      acc[j] += vf * sfw[j * KC + h] + vb * sfw[j * KC + Hh + h];
  }
#pragma unroll
  for (int j = 0; j < Tt; ++j) logits[(size_t)pos * Tt + j] = acc[j];
}

// ---------------- CRF score + forward algorithm + loss ----------------
__global__ __launch_bounds__(64)
void k_crf(const float* __restrict__ logits, const int* __restrict__ label,
           const int* __restrict__ seqlen,
           const float* __restrict__ start_t, const float* __restrict__ end_t,
           const float* __restrict__ trans, float* __restrict__ out) {
  __shared__ float red[Bb];
  const int b = threadIdx.x;

  float st[Tt], en[Tt], tr[Tt * Tt], alpha[Tt];
#pragma unroll
  for (int i = 0; i < Tt; ++i) { st[i] = start_t[i]; en[i] = end_t[i]; }
  for (int i = 0; i < Tt * Tt; ++i) tr[i] = trans[i];

  const int L = seqlen[b];
  int tag_prev = label[b * Ss + 0];
  const float* lg0 = logits + (size_t)b * Tt;   // t = 0
  float score = st[tag_prev] + lg0[tag_prev];
#pragma unroll
  for (int j = 0; j < Tt; ++j) alpha[j] = st[j] + lg0[j];

  for (int t = 1; t < Ss; ++t) {
    const int tag = label[b * Ss + t];
    if (t < L) {
      const float* lg = logits + ((size_t)t * Bb + b) * Tt;
      score += tr[tag_prev * Tt + tag] + lg[tag];
      float na[Tt];
#pragma unroll
      for (int j = 0; j < Tt; ++j) {
        float m = -3.4e38f;
#pragma unroll
        for (int i = 0; i < Tt; ++i) m = fmaxf(m, alpha[i] + tr[i * Tt + j]);
        float s = 0.0f;
#pragma unroll
        for (int i = 0; i < Tt; ++i) s += __expf(alpha[i] + tr[i * Tt + j] - m);
        na[j] = m + __logf(s) + lg[j];
      }
#pragma unroll
      for (int j = 0; j < Tt; ++j) alpha[j] = na[j];
    }
    tag_prev = tag;
  }
  score += en[label[b * Ss + (L - 1)]];

  float m = -3.4e38f;
#pragma unroll
  for (int j = 0; j < Tt; ++j) m = fmaxf(m, alpha[j] + en[j]);
  float s = 0.0f;
#pragma unroll
  for (int j = 0; j < Tt; ++j) s += __expf(alpha[j] + en[j] - m);
  red[b] = (m + __logf(s)) - score;

  __syncthreads();
  if (b == 0) {
    float tot = 0.0f;
    for (int i = 0; i < Bb; ++i) tot += red[i];
    out[0] = tot;
  }
}

// ---------------- launcher ----------------
extern "C" void kernel_launch(void* const* d_in, const int* in_sizes, int n_in,
                              void* d_out, int out_size, void* d_ws, size_t ws_size,
                              hipStream_t stream) {
  (void)in_sizes; (void)n_in; (void)out_size; (void)ws_size;
  const int*   x      = (const int*)  d_in[0];
  const int*   seqlen = (const int*)  d_in[1];
  const int*   label  = (const int*)  d_in[2];
  const float* emb    = (const float*)d_in[3];
  const float* wihf   = (const float*)d_in[4];
  const float* whhf   = (const float*)d_in[5];
  const float* bihf   = (const float*)d_in[6];
  const float* bhhf   = (const float*)d_in[7];
  const float* wihb   = (const float*)d_in[8];
  const float* whhb   = (const float*)d_in[9];
  const float* bihb   = (const float*)d_in[10];
  const float* bhhb   = (const float*)d_in[11];
  const float* fcw    = (const float*)d_in[12];
  const float* fcb    = (const float*)d_in[13];
  const float* start_t= (const float*)d_in[14];
  const float* end_t  = (const float*)d_in[15];
  const float* trans  = (const float*)d_in[16];
  float* out = (float*)d_out;

  // workspace layout (bytes)
  char* ws = (char*)d_ws;
  size_t off = 0;
  bf16* xsb   = (bf16*)(ws + off); off += (size_t)Ss * Bb * Hh * 2;   // 16 MB
  bf16* wbp   = (bf16*)(ws + off); off += (size_t)2 * G4H * KC * 2;   //  2 MB
  float* bc   = (float*)(ws + off); off += (size_t)2 * G4H * 4;       //  8 KB
  bf16* hsf   = (bf16*)(ws + off); off += (size_t)Ss * Bb * Hh * 2;   // 16 MB
  bf16* hsb   = (bf16*)(ws + off); off += (size_t)Ss * Bb * Hh * 2;   // 16 MB
  float* logits = (float*)(ws + off); off += (size_t)Ss * Bb * Tt * 4;//  1.1 MB

  k_embed <<<(Ss * Bb * Hh + 255) / 256, 256, 0, stream>>>(x, emb, xsb);
  k_pack_w<<<(2 * G4H * KC + 255) / 256, 256, 0, stream>>>(wihf, whhf, wihb, whhb, wbp);
  k_bias  <<<(2 * G4H + 255) / 256, 256, 0, stream>>>(bihf, bhhf, bihb, bhhb, bc);

  const size_t lstm_smem = (size_t)64 * 1024 * 4 + (size_t)64 * 256 * 2; // 288 KB
  k_lstm  <<<2, 1024, lstm_smem, stream>>>(xsb, wbp, bc, hsf, hsb);

  k_logits<<<(Ss * Bb + 255) / 256, 256, 0, stream>>>(hsf, hsb, fcw, fcb, logits);
  k_crf   <<<1, 64, 0, stream>>>(logits, label, seqlen, start_t, end_t, trans, out);
}
